// DecoderBlock_17300128268959
// MI455X (gfx1250) — compile-verified, matrix operand order checked by reading, not run on hardware
//
#include <hip/hip_runtime.h>
#include <hip/hip_bf16.h>
#include <math.h>

// ---------------------------------------------------------------------------
// CDNA5 (gfx1250) decoder block: fp32 residual stream, bf16 WMMA GEMMs,
// async global->LDS weight streaming, all fragment loads row-contiguous.
// ---------------------------------------------------------------------------

typedef __attribute__((ext_vector_type(16))) __bf16 v16bf;
typedef __attribute__((ext_vector_type(8)))  float  v8f;

// --------------------------- async LDS copy --------------------------------
#if __has_builtin(__builtin_amdgcn_global_load_async_to_lds_b128)
#define ASYNC_LDS 1
typedef int v4i __attribute__((vector_size(4 * sizeof(int))));
typedef __attribute__((address_space(1))) v4i GV4i;  // global (device) AS
typedef __attribute__((address_space(3))) v4i LV4i;  // LDS AS
__device__ __forceinline__ void async_copy_b128(const void* g, void* l) {
  __builtin_amdgcn_global_load_async_to_lds_b128(
      (GV4i*)(unsigned long long)(size_t)g,
      (LV4i*)(unsigned int)(size_t)l, 0, 0);
}
__device__ __forceinline__ void wait_async_lds() {
  asm volatile("s_wait_asynccnt 0x0" ::: "memory");
}
#else
#define ASYNC_LDS 0
__device__ __forceinline__ void async_copy_b128(const void* g, void* l) {
  *(uint4*)l = *(const uint4*)g;
}
__device__ __forceinline__ void wait_async_lds() {}
#endif

// --------------------------- WMMA fragments --------------------------------
// K index inside a 16-bit 16x32 fragment for element e (0..15), lane half hi,
// per CDNA5 ISA 7.12.2.
__device__ __forceinline__ int frag_k(int e, int hi) {
  return (e < 8) ? (hi * 8 + e) : (16 + hi * 8 + (e - 8));
}

// Row-contiguous fragment: s[row/col = lane&15][contraction].  Used for A
// operands and for B operands staged transposed (contraction contiguous).
// All accesses are 16B-aligned groups of 8 bf16 -> ds_load_b128.
__device__ __forceinline__ v16bf load_frag(const __bf16* s, int ld, int lane) {
  const int m = lane & 15, hi = lane >> 4;
  v16bf a;
#pragma unroll
  for (int e = 0; e < 16; ++e) a[e] = s[m * ld + frag_k(e, hi)];
  return a;
}

__device__ __forceinline__ v8f wmma_bf16(v16bf a, v16bf b, v8f c) {
  return __builtin_amdgcn_wmma_f32_16x16x32_bf16(false, a, false, b, (short)0, c,
                                                 false, false);
}

// ---------------------------------------------------------------------------
// Weight prep: W[K][N] f32 -> Wt[N][K] bf16 (transpose + convert), tiled.
// One-off ~96 MB of traffic per call, amortized over ~170 GFLOP of WMMA.
// ---------------------------------------------------------------------------
__global__ __launch_bounds__(256) void transpose_cvt_kernel(
    const float* __restrict__ src, __bf16* __restrict__ dst, int K, int N) {
  __shared__ float tile[32][33];
  const int k0 = blockIdx.y * 32, n0 = blockIdx.x * 32;
  const int tx = threadIdx.x & 31, ty = threadIdx.x >> 5;  // 8 row-groups
#pragma unroll
  for (int r = ty; r < 32; r += 8)
    tile[r][tx] = src[(size_t)(k0 + r) * N + (n0 + tx)];
  __syncthreads();
#pragma unroll
  for (int r = ty; r < 32; r += 8)
    dst[(size_t)(n0 + r) * K + (k0 + tx)] = (__bf16)tile[tx][r];
}

// ---------------------------------------------------------------------------
// GEMM: out[M,N] = fuse(A[M,K] @ W[K,N] + bias), W given transposed bf16
// Wt[N][K].  256 threads = 8 waves, tile 128(M) x 64(N), K-step 32, LDS
// double-buffered.  Weights stream via async global->LDS b128 (no VGPRs);
// activations converted f32->bf16 through VGPRs.  One barrier per K-step.
// FUSE: 0=none, 1=clipped gelu, 2=+residual.
// ---------------------------------------------------------------------------
template <int FUSE>
__global__ __launch_bounds__(256) void gemm_wmma_kernel(
    const float* __restrict__ A, const __bf16* __restrict__ Wt,
    const float* __restrict__ bias, const float* __restrict__ res,
    float* __restrict__ out, int M, int N, int K) {
  __shared__ __bf16 sA[2][128][40];  // [m][k], stride 80B (16B aligned)
  __shared__ __bf16 sB[2][64][40];   // [n][k] (pre-transposed weights)
  const int tid = threadIdx.x, lane = tid & 31, wave = tid >> 5;
  const int m0 = blockIdx.y * 128, n0 = blockIdx.x * 64;

  auto stage_B = [&](int buf, int k0) {  // 64x32 bf16: one async b128/thread
    const int n = tid >> 2, c = (tid & 3) * 8;
    async_copy_b128(Wt + (size_t)(n0 + n) * K + (k0 + c), &sB[buf][n][c]);
  };
  auto stage_A = [&](int buf, int k0) {  // 128x32 f32 -> bf16, float4 loads
#pragma unroll
    for (int e = 0; e < 4; ++e) {
      const int idx = tid + e * 256;
      const int r = idx >> 3, c = (idx & 7) * 4;
      const float4 v = *(const float4*)&A[(size_t)(m0 + r) * K + (k0 + c)];
      sA[buf][r][c + 0] = (__bf16)v.x;
      sA[buf][r][c + 1] = (__bf16)v.y;
      sA[buf][r][c + 2] = (__bf16)v.z;
      sA[buf][r][c + 3] = (__bf16)v.w;
    }
  };

  stage_B(0, 0);
  stage_A(0, 0);
  v8f acc[4] = {};
  int buf = 0;
  for (int k0 = 0; k0 < K; k0 += 32, buf ^= 1) {
    wait_async_lds();   // this wave's async B transfers complete
    __syncthreads();    // all waves: A stores visible + B tiles resident
    if (k0 + 32 < K) {
      stage_B(buf ^ 1, k0 + 32);
      stage_A(buf ^ 1, k0 + 32);
      __builtin_prefetch(&A[(size_t)(m0 + (tid >> 1)) * K + (k0 + 64)], 0, 0);
    }
    const v16bf a = load_frag(&sA[buf][wave * 16][0], 40, lane);
#pragma unroll
    for (int j = 0; j < 4; ++j) {
      const v16bf b = load_frag(&sB[buf][j * 16][0], 40, lane);
      acc[j] = wmma_bf16(a, b, acc[j]);
    }
  }

  const int hi = lane >> 4, n = lane & 15;
#pragma unroll
  for (int j = 0; j < 4; ++j) {
#pragma unroll
    for (int r = 0; r < 8; ++r) {
      const int row = m0 + wave * 16 + r + 8 * hi;  // C/D layout: M = r + 8*hi
      const int col = n0 + j * 16 + n;
      float v = acc[j][r] + bias[col];
      if (FUSE == 1) {  // safe_gelu: clip(+-6) -> tanh-gelu -> clip(+-6)
        v = fminf(fmaxf(v, -6.f), 6.f);
        const float g =
            0.5f * v * (1.f + tanhf(0.7978845608f * (v + 0.044715f * v * v * v)));
        v = fminf(fmaxf(g, -6.f), 6.f);
      }
      if (FUSE == 2) v += res[(size_t)row * N + col];
      out[(size_t)row * N + col] = v;
    }
  }
}

// ---------------------------------------------------------------------------
// LayerNorm / RMSNorm: one block per row.
// ---------------------------------------------------------------------------
__global__ __launch_bounds__(256) void layernorm_kernel(
    const float* __restrict__ x, const float* __restrict__ w,
    const float* __restrict__ b, float* __restrict__ out, int C) {
  const size_t row = blockIdx.x;
  const int tid = threadIdx.x;
  const float* xr = x + row * C;
  __shared__ float red[256];
  float s = 0.f;
  for (int i = tid; i < C; i += 256) s += xr[i];
  red[tid] = s;
  __syncthreads();
  for (int o = 128; o > 0; o >>= 1) {
    if (tid < o) red[tid] += red[tid + o];
    __syncthreads();
  }
  const float mean = red[0] * (1.f / C);
  __syncthreads();
  float v = 0.f;
  for (int i = tid; i < C; i += 256) {
    const float d = xr[i] - mean;
    v += d * d;
  }
  red[tid] = v;
  __syncthreads();
  for (int o = 128; o > 0; o >>= 1) {
    if (tid < o) red[tid] += red[tid + o];
    __syncthreads();
  }
  const float inv = rsqrtf(red[0] * (1.f / C) + 1e-5f);
  for (int i = tid; i < C; i += 256)
    out[row * C + i] = (xr[i] - mean) * inv * w[i] + b[i];
}

__global__ __launch_bounds__(256) void rmsnorm_kernel(float* __restrict__ x,
                                                      const float* __restrict__ g,
                                                      int C) {
  const size_t row = blockIdx.x;
  const int tid = threadIdx.x;
  float* xr = x + row * C;
  __shared__ float red[256];
  float s = 0.f;
  for (int i = tid; i < C; i += 256) {
    const float v = xr[i];
    s += v * v;
  }
  red[tid] = s;
  __syncthreads();
  for (int o = 128; o > 0; o >>= 1) {
    if (tid < o) red[tid] += red[tid + o];
    __syncthreads();
  }
  const float inv = rsqrtf(red[0] * (1.f / C) + 1e-6f);
  for (int i = tid; i < C; i += 256) xr[i] = xr[i] * inv * g[i];
}

// ---------------------------------------------------------------------------
// RoPE on q and k in-place, theta = 50000, head dim 64.
// ---------------------------------------------------------------------------
__global__ __launch_bounds__(256) void rope_kernel(float* __restrict__ q,
                                                   float* __restrict__ k, int B,
                                                   int T, int H, int Dh) {
  const int pairs = Dh / 2;
  const int total = B * T * H * pairs;
  const int idx = blockIdx.x * blockDim.x + threadIdx.x;
  if (idx >= total) return;
  const int i = idx % pairs;
  const int h = (idx / pairs) % H;
  const int t = (idx / (pairs * H)) % T;
  const int b = idx / (pairs * H * T);
  const float inv_freq = __expf(-((float)(2 * i) / (float)Dh) * __logf(50000.0f));
  float sn, cs;
  __sincosf((float)t * inv_freq, &sn, &cs);
  const size_t base = ((size_t)(b * T + t) * (H * Dh)) + h * Dh + 2 * i;
  const float qr = q[base], qi = q[base + 1];
  q[base] = qr * cs - qi * sn;
  q[base + 1] = qr * sn + qi * cs;
  const float kr = k[base], ki = k[base + 1];
  k[base] = kr * cs - ki * sn;
  k[base + 1] = kr * sn + ki * cs;
}

// ---------------------------------------------------------------------------
// Flash attention with WMMA.  Block = 64 q-rows x (b,h); 4 waves, each wave
// owns 16 query rows.  Keys in chunks of 32 with online softmax.  V staged
// transposed so the P.V B-fragment is contraction-contiguous (b128 loads);
// P round-trips through a per-wave LDS buffer guarded by s_wait_dscnt.
// ---------------------------------------------------------------------------
__global__ __launch_bounds__(128) void attn_wmma_kernel(
    const float* __restrict__ Q, const float* __restrict__ Kx,
    const float* __restrict__ V, const unsigned char* __restrict__ mask,
    float* __restrict__ O, int B, int T, int H, int Dh) {
  const int bh = blockIdx.y;
  const int b = bh / H, h = bh % H;
  const int tid = threadIdx.x, lane = tid & 31, wave = tid >> 5;
  const int C = H * Dh;
  const int q0 = blockIdx.x * 64;
  const int hi = lane >> 4, ln = lane & 15;

  __shared__ __bf16 sQ[64][72];    // [q][d]
  __shared__ __bf16 sK[32][72];    // [key][d]    (QK^T: d-contiguous)
  __shared__ __bf16 sVt[64][40];   // [d][key]    (P.V: key-contiguous)
  __shared__ __bf16 sP[4][16][40];
  __shared__ float km[32];

#pragma unroll
  for (int i = tid; i < 64 * 64; i += 128) {
    const int r = i >> 6, c = i & 63;
    sQ[r][c] = (__bf16)Q[(size_t)(b * T + q0 + r) * C + h * Dh + c];
  }

  float mrow[8], lrow[8];
  v8f o[4] = {};
#pragma unroll
  for (int r = 0; r < 8; ++r) {
    mrow[r] = -3.0e38f;
    lrow[r] = 0.f;
  }
  __syncthreads();

  for (int kc = 0; kc < T; kc += 32) {
    __syncthreads();
#pragma unroll
    for (int i = tid; i < 32 * 64; i += 128) {
      const int r = i >> 6, c = i & 63;
      const float kv = Kx[(size_t)(b * T + kc + r) * C + h * Dh + c];
      const float vv = V[(size_t)(b * T + kc + r) * C + h * Dh + c];
      sK[r][c] = (__bf16)kv;
      sVt[c][r] = (__bf16)vv;
    }
    if (tid < 32) km[tid] = mask[(size_t)b * T + kc + tid] ? 1.f : 0.f;
    __syncthreads();

    // S = Q * K^T: 2 key sub-tiles x 2 contraction steps of 32.
    v8f s[2] = {};
#pragma unroll
    for (int j = 0; j < 2; ++j) {
#pragma unroll
      for (int ks = 0; ks < 64; ks += 32) {
        const v16bf a = load_frag(&sQ[wave * 16][ks], 72, lane);
        const v16bf bt = load_frag(&sK[j * 16][ks], 72, lane);
        s[j] = wmma_bf16(a, bt, s[j]);
      }
    }

    const float scale = 0.125f;  // 1/sqrt(64)
    const float km0 = km[ln], km1 = km[16 + ln];
    float pv0[8], pv1[8], alpha[8];
#pragma unroll
    for (int r = 0; r < 8; ++r) {
      float v0 = km0 > 0.f ? s[0][r] * scale : -3.0e38f;
      float v1 = km1 > 0.f ? s[1][r] * scale : -3.0e38f;
      float mx = fmaxf(v0, v1);
#pragma unroll
      for (int off = 8; off > 0; off >>= 1) mx = fmaxf(mx, __shfl_xor(mx, off, 16));
      const float mnew = fmaxf(mrow[r], mx);
      const float a0 = __expf(v0 - mnew) * km0;
      const float a1 = __expf(v1 - mnew) * km1;
      float rs = a0 + a1;
#pragma unroll
      for (int off = 8; off > 0; off >>= 1) rs += __shfl_xor(rs, off, 16);
      const float al = __expf(mrow[r] - mnew);
      lrow[r] = lrow[r] * al + rs;
      mrow[r] = mnew;
      alpha[r] = al;
      pv0[r] = a0;
      pv1[r] = a1;
    }
#pragma unroll
    for (int nt = 0; nt < 4; ++nt)
#pragma unroll
      for (int r = 0; r < 8; ++r) o[nt][r] *= alpha[r];

    // P: D-layout -> per-wave LDS -> A-fragment (same-wave LDS ordering).
#pragma unroll
    for (int r = 0; r < 8; ++r) {
      sP[wave][r + 8 * hi][ln] = (__bf16)pv0[r];
      sP[wave][r + 8 * hi][16 + ln] = (__bf16)pv1[r];
    }
    asm volatile("s_wait_dscnt 0x0" ::: "memory");
    const v16bf pa = load_frag(&sP[wave][0][0], 40, lane);
#pragma unroll
    for (int nt = 0; nt < 4; ++nt) {
      const v16bf bv = load_frag(&sVt[nt * 16][0], 40, lane);
      o[nt] = wmma_bf16(pa, bv, o[nt]);
    }
  }

#pragma unroll
  for (int nt = 0; nt < 4; ++nt) {
#pragma unroll
    for (int r = 0; r < 8; ++r) {
      const int row = q0 + wave * 16 + r + 8 * hi;
      const float den = lrow[r];
      const float val = den > 0.f ? o[nt][r] / den : 0.f;
      O[(size_t)(b * T + row) * C + h * Dh + nt * 16 + ln] = val;
    }
  }
}

// ---------------------------------------------------------------------------
// Fourier layer: per (b, channel) series of length T.  Direct DFT over bins
// 1..511, shared-mem top-6 by magnitude, cosine reconstruction; writes final
// output = x + fourier(x).
// ---------------------------------------------------------------------------
__global__ __launch_bounds__(256) void fourier_kernel(const float* __restrict__ X,
                                                      float* __restrict__ out,
                                                      int T, int C, int topk) {
  const int b = blockIdx.x / C, d = blockIdx.x % C;
  const int tid = threadIdx.x;
  __shared__ float xloc[1024];
  __shared__ float sre[512], sim[512], smag[512];
  __shared__ float rmax[256];
  __shared__ int rarg[256];
  __shared__ int topi[8];

  for (int t = tid; t < T; t += 256) xloc[t] = X[(size_t)(b * T + t) * C + d];
  __syncthreads();

  for (int f = tid; f < 512; f += 256) {
    if (f == 0) {
      sre[0] = 0.f; sim[0] = 0.f; smag[0] = -1.f;  // bin 0 excluded
      continue;
    }
    const float w = 6.28318530718f * (float)f / (float)T;
    float re = 0.f, im = 0.f;
    for (int t = 0; t < T; ++t) {
      float sn, cs;
      __sincosf(w * (float)t, &sn, &cs);
      re += xloc[t] * cs;
      im -= xloc[t] * sn;
    }
    sre[f] = re;
    sim[f] = im;
    smag[f] = re * re + im * im;
  }
  __syncthreads();

  for (int k = 0; k < topk; ++k) {
    float bv = smag[tid];
    int bi = tid;
    const float v2 = smag[tid + 256];
    if (v2 > bv) { bv = v2; bi = tid + 256; }
    rmax[tid] = bv;
    rarg[tid] = bi;
    __syncthreads();
    for (int o = 128; o > 0; o >>= 1) {
      if (tid < o && rmax[tid + o] > rmax[tid]) {
        rmax[tid] = rmax[tid + o];
        rarg[tid] = rarg[tid + o];
      }
      __syncthreads();
    }
    if (tid == 0) {
      topi[k] = rarg[0];
      smag[rarg[0]] = -2.f;
    }
    __syncthreads();
  }

  for (int t = tid; t < T; t += 256) {
    float acc = 0.f;
    for (int k = 0; k < topk; ++k) {
      const int f = topi[k];
      const float re = sre[f], im = sim[f];
      const float amp = sqrtf(re * re + im * im);
      const float ph = atan2f(im, re);
      acc += 2.f * amp * __cosf(6.28318530718f * (float)f * (float)t / (float)T + ph);
    }
    out[(size_t)(b * T + t) * C + d] = xloc[t] + acc;
  }
}

// ---------------------------------------------------------------------------
// Host-side pipeline.
// ---------------------------------------------------------------------------
extern "C" void kernel_launch(void* const* d_in, const int* in_sizes, int n_in,
                              void* d_out, int out_size, void* d_ws, size_t ws_size,
                              hipStream_t stream) {
  (void)in_sizes; (void)n_in; (void)out_size; (void)ws_size;
  const int B = 4, T = 1024, C = 1024, DFF = 4096, H = 16, Dh = 64;
  const int M = B * T;

  const float* x = (const float*)d_in[0];
  const float* enc = (const float*)d_in[1];
  const unsigned char* mask = (const unsigned char*)d_in[2];
  const float* ln1w = (const float*)d_in[3];
  const float* ln1b = (const float*)d_in[4];
  const float* ln2w = (const float*)d_in[5];
  const float* ln2b = (const float*)d_in[6];
  const float* ln3w = (const float*)d_in[7];
  const float* ln3b = (const float*)d_in[8];
  const float* sa_wq = (const float*)d_in[9];
  const float* sa_bq = (const float*)d_in[10];
  const float* sa_wk = (const float*)d_in[11];
  const float* sa_bk = (const float*)d_in[12];
  const float* sa_wv = (const float*)d_in[13];
  const float* sa_bv = (const float*)d_in[14];
  const float* sa_wo = (const float*)d_in[15];
  const float* sa_bo = (const float*)d_in[16];
  const float* sa_qn = (const float*)d_in[17];
  const float* sa_kn = (const float*)d_in[18];
  const float* ca_wq = (const float*)d_in[19];
  const float* ca_bq = (const float*)d_in[20];
  const float* ca_wk = (const float*)d_in[21];
  const float* ca_bk = (const float*)d_in[22];
  const float* ca_wv = (const float*)d_in[23];
  const float* ca_bv = (const float*)d_in[24];
  const float* ca_wo = (const float*)d_in[25];
  const float* ca_bo = (const float*)d_in[26];
  const float* ca_qn = (const float*)d_in[27];
  const float* ca_kn = (const float*)d_in[28];
  const float* w1 = (const float*)d_in[29];
  const float* b1 = (const float*)d_in[30];
  const float* w2 = (const float*)d_in[31];
  const float* b2 = (const float*)d_in[32];

  char* ws = (char*)d_ws;
  const size_t SZ = (size_t)M * C * sizeof(float);  // 16 MiB
  float* h = (float*)(ws + 0 * SZ);
  float* qb = (float*)(ws + 1 * SZ);
  float* kb = (float*)(ws + 2 * SZ);
  float* vb = (float*)(ws + 3 * SZ);
  float* ab = (float*)(ws + 4 * SZ);
  float* r0 = (float*)(ws + 5 * SZ);
  float* r1 = (float*)(ws + 6 * SZ);
  float* hid = (float*)(ws + 7 * SZ);              // M x DFF f32, 64 MiB
  char* wbase = ws + 7 * SZ + (size_t)M * DFF * 4; // bf16 weight arena
  const size_t WSQ = (size_t)C * C * sizeof(__bf16);  // 2 MiB
  __bf16* sa_wqT = (__bf16*)(wbase + 0 * WSQ);
  __bf16* sa_wkT = (__bf16*)(wbase + 1 * WSQ);
  __bf16* sa_wvT = (__bf16*)(wbase + 2 * WSQ);
  __bf16* sa_woT = (__bf16*)(wbase + 3 * WSQ);
  __bf16* ca_wqT = (__bf16*)(wbase + 4 * WSQ);
  __bf16* ca_wkT = (__bf16*)(wbase + 5 * WSQ);
  __bf16* ca_wvT = (__bf16*)(wbase + 6 * WSQ);
  __bf16* ca_woT = (__bf16*)(wbase + 7 * WSQ);
  __bf16* w1T = (__bf16*)(wbase + 8 * WSQ);                  // [DFF][C]
  __bf16* w2T = (__bf16*)(wbase + 8 * WSQ + (size_t)C * DFF * 2);  // [C][DFF]

  const dim3 blk128(128), blk256(256);
  const dim3 gN(C / 64, M / 128);    // GEMMs with N = C
  const dim3 gF(DFF / 64, M / 128);  // GEMM with N = DFF
  const dim3 gAttn(T / 64, B * H);
  const dim3 gTs(C / 32, C / 32);          // square weight transpose
  const dim3 gT1(DFF / 32, C / 32);        // w1: [C][DFF] -> [DFF][C]
  const dim3 gT2(C / 32, DFF / 32);        // w2: [DFF][C] -> [C][DFF]

  // ---- 0) weight prep: transpose + f32->bf16 once per call ----
  transpose_cvt_kernel<<<gTs, blk256, 0, stream>>>(sa_wq, sa_wqT, C, C);
  transpose_cvt_kernel<<<gTs, blk256, 0, stream>>>(sa_wk, sa_wkT, C, C);
  transpose_cvt_kernel<<<gTs, blk256, 0, stream>>>(sa_wv, sa_wvT, C, C);
  transpose_cvt_kernel<<<gTs, blk256, 0, stream>>>(sa_wo, sa_woT, C, C);
  transpose_cvt_kernel<<<gTs, blk256, 0, stream>>>(ca_wq, ca_wqT, C, C);
  transpose_cvt_kernel<<<gTs, blk256, 0, stream>>>(ca_wk, ca_wkT, C, C);
  transpose_cvt_kernel<<<gTs, blk256, 0, stream>>>(ca_wv, ca_wvT, C, C);
  transpose_cvt_kernel<<<gTs, blk256, 0, stream>>>(ca_wo, ca_woT, C, C);
  transpose_cvt_kernel<<<gT1, blk256, 0, stream>>>(w1, w1T, C, DFF);
  transpose_cvt_kernel<<<gT2, blk256, 0, stream>>>(w2, w2T, DFF, C);

  // ---- 1) self-attention: x1 = x + sa(LN1(x)) ----
  layernorm_kernel<<<M, blk256, 0, stream>>>(x, ln1w, ln1b, h, C);
  gemm_wmma_kernel<0><<<gN, blk256, 0, stream>>>(h, sa_wqT, sa_bq, nullptr, qb, M, C, C);
  rmsnorm_kernel<<<M, blk256, 0, stream>>>(qb, sa_qn, C);
  gemm_wmma_kernel<0><<<gN, blk256, 0, stream>>>(h, sa_wkT, sa_bk, nullptr, kb, M, C, C);
  rmsnorm_kernel<<<M, blk256, 0, stream>>>(kb, sa_kn, C);
  gemm_wmma_kernel<0><<<gN, blk256, 0, stream>>>(h, sa_wvT, sa_bv, nullptr, vb, M, C, C);
  attn_wmma_kernel<<<gAttn, blk128, 0, stream>>>(qb, kb, vb, mask, ab, B, T, H, Dh);
  gemm_wmma_kernel<2><<<gN, blk256, 0, stream>>>(ab, sa_woT, sa_bo, x, r0, M, C, C);

  // ---- 2) cross-attention with RoPE: x2 = x1 + ca(LN2(x1), enc) ----
  layernorm_kernel<<<M, blk256, 0, stream>>>(r0, ln2w, ln2b, h, C);
  gemm_wmma_kernel<0><<<gN, blk256, 0, stream>>>(h, ca_wqT, ca_bq, nullptr, qb, M, C, C);
  rmsnorm_kernel<<<M, blk256, 0, stream>>>(qb, ca_qn, C);
  gemm_wmma_kernel<0><<<gN, blk256, 0, stream>>>(enc, ca_wkT, ca_bk, nullptr, kb, M, C, C);
  rmsnorm_kernel<<<M, blk256, 0, stream>>>(kb, ca_kn, C);
  gemm_wmma_kernel<0><<<gN, blk256, 0, stream>>>(enc, ca_wvT, ca_bv, nullptr, vb, M, C, C);
  {
    const int total = B * T * H * (Dh / 2);
    rope_kernel<<<(total + 255) / 256, blk256, 0, stream>>>(qb, kb, B, T, H, Dh);
  }
  attn_wmma_kernel<<<gAttn, blk128, 0, stream>>>(qb, kb, vb, mask, ab, B, T, H, Dh);
  gemm_wmma_kernel<2><<<gN, blk256, 0, stream>>>(ab, ca_woT, ca_bo, r0, r1, M, C, C);

  // ---- 3) MLP: x3 = x2 + W2(gelu(W1 LN3(x2))) ----
  layernorm_kernel<<<M, blk256, 0, stream>>>(r1, ln3w, ln3b, h, C);
  gemm_wmma_kernel<1><<<gF, blk256, 0, stream>>>(h, w1T, b1, nullptr, hid, M, DFF, C);
  gemm_wmma_kernel<2><<<gN, blk256, 0, stream>>>(hid, w2T, b2, r1, r0, M, C, DFF);

  // ---- 4) fourier layer: out = x3 + fourier(x3) ----
  fourier_kernel<<<B * C, blk256, 0, stream>>>(r0, (float*)d_out, T, C, /*topk=*/6);
}
